// UV_Aggregator_51196010168833
// MI455X (gfx1250) — matrix operand order repaired and checked by reading, not compiled, not run on hardware
//
#include <hip/hip_runtime.h>
#include <hip/hip_bf16.h>
#include <math.h>

typedef __attribute__((ext_vector_type(16))) _Float16 v16h;
typedef __attribute__((ext_vector_type(8)))  float    v8f;
typedef int v4i_t __attribute__((vector_size(16)));   // async-to-LDS b128 ptr type

#define XS   132   // xs row stride in floats (128 + 4 pad -> bank stagger)
#define OS   66    // o_hist row stride in floats (64 + 2 pad)
#define LNEI 50
#define DD   64

// f16 weight cache layout inside d_ws (halves):
//   [0,8192)      w_r1  (64x128)
//   [8192,12288)  w_r2  (64x64)
//   [12288,20480) att1  (64x128)
//   [20480,24576) att2  (64x64)
#define WH_TOTAL 24576

#if __has_builtin(__builtin_amdgcn_global_load_async_to_lds_b128)
#define HAVE_ASYNC_LDS 1
#endif

__device__ __forceinline__ void wait_async0() {
#if defined(HAVE_ASYNC_LDS)
#if __has_builtin(__builtin_amdgcn_s_wait_asynccnt)
  __builtin_amdgcn_s_wait_asynccnt(0);
#else
  asm volatile("s_wait_asynccnt 0" ::: "memory");
#endif
#endif
}

// A fragment (16x32 f16, this lane's row) from one or two 64-wide f32 LDS rows.
// Layout: halves[0..15]=K k0+2*off+0..15 (contiguous, matches B fragment)
// with off = (lane>=16) ? 8 : 0.
__device__ __forceinline__ v16h load_a_frag(const float* lo, const float* hi64,
                                            int k0, int off) {
  v16h a;
#pragma unroll
  for (int i = 0; i < 8; ++i) {
    int k1 = k0 + 2 * off + i;
    int k2 = k0 + 2 * off + 8 + i;
    float f1 = (k1 < 64) ? lo[k1] : hi64[k1 - 64];
    float f2 = (k2 < 64) ? lo[k2] : hi64[k2 - 64];
    a[i]     = (_Float16)f1;
    a[i + 8] = (_Float16)f2;
  }
  return a;
}

// B fragment (32x16 f16): B[k][n] = W[n][k], W is [64][K] row-major.
// lanes0-15: N=lane, K=k0..k0+15 ; lanes16-31: N=lane-16, K=k0+16..k0+31.
// For pre-converted f16 weights this is one aligned 32B vector load.
template<typename WT>
__device__ __forceinline__ v16h load_b_frag(const WT* __restrict__ W, int K,
                                            int nt, int k0, int lane) {
  int n  = nt * 16 + (lane & 15);
  int kb = k0 + ((lane & 16) ? 16 : 0);
  if constexpr (sizeof(WT) == 2) {
    return *reinterpret_cast<const v16h*>(W + n * K + kb);
  } else {
    const WT* p = W + n * K + kb;
    v16h bf;
#pragma unroll
    for (int i = 0; i < 16; ++i) bf[i] = (_Float16)p[i];
    return bf;
  }
}

// One wave computes a 16x64 tile: dst = relu(A[16xK] * W^T + bias).
template<int K, typename WT>
__device__ __forceinline__ void gemm16x64(const float* lo, const float* hi64,
                                          const WT* __restrict__ W,
                                          const float* __restrict__ bias,
                                          float* dst, int dstride, int dcol0,
                                          int lane) {
  const int off = (lane & 16) ? 8 : 0;
  v16h A[K / 32];
#pragma unroll
  for (int kt = 0; kt < K / 32; ++kt) A[kt] = load_a_frag(lo, hi64, kt * 32, off);
  asm volatile("" ::: "memory");  // A reads must precede aliasing LDS stores
#pragma unroll
  for (int nt = 0; nt < 4; ++nt) {
    v8f acc = {};
#pragma unroll
    for (int kt = 0; kt < K / 32; ++kt) {
      v16h Bf = load_b_frag<WT>(W, K, nt, kt * 32, lane);
      acc = __builtin_amdgcn_wmma_f32_16x16x32_f16(false, A[kt], false, Bf,
                                                   (short)0, acc, false, false);
    }
    int col = nt * 16 + (lane & 15);
    float bb = bias[col];
#pragma unroll
    for (int r = 0; r < 8; ++r) {
      int row = r + ((lane & 16) ? 8 : 0);
      dst[row * dstride + dcol0 + col] = fmaxf(acc[r] + bb, 0.f);
    }
  }
}

// one-shot f32 -> f16 weight conversion into d_ws (weights are block-invariant)
__global__ void __launch_bounds__(256)
cvt_weights_kernel(const float* __restrict__ w1, const float* __restrict__ w2,
                   const float* __restrict__ a1, const float* __restrict__ a2,
                   _Float16* __restrict__ out) {
  int i = blockIdx.x * 256 + threadIdx.x;   // 0..24575
  float v;
  if (i < 8192)        v = w1[i];
  else if (i < 12288)  v = w2[i - 8192];
  else if (i < 20480)  v = a1[i - 12288];
  else                 v = a2[i - 20480];
  out[i] = (_Float16)v;
}

template<typename WT>
__global__ void __launch_bounds__(128)
uv_agg_kernel(const int* __restrict__ nodes,
              const int* __restrict__ history_uv,
              const int* __restrict__ history_r,
              const float* __restrict__ v2e,
              const float* __restrict__ u2e,
              const float* __restrict__ r2e,
              const WT* __restrict__ w1w, const float* __restrict__ w1b,
              const WT* __restrict__ w2w, const float* __restrict__ w2b,
              const WT* __restrict__ a1w, const float* __restrict__ a1b,
              const WT* __restrict__ a2w, const float* __restrict__ a2b,
              const float* __restrict__ a3w, const float* __restrict__ a3b,
              float* __restrict__ out) {
  __shared__ float xs[4 * 16 * XS];   // per-wave activation scratch (128 wide)
  __shared__ float oh[64 * OS];       // o_history, f32, rows = padded L
  __shared__ float lgt[64];           // attention logits
  __shared__ float uv_sh[64];         // u2e[nodes[b]]

  const int  b    = blockIdx.x;
  const int  tid  = threadIdx.x;
  const int  w    = tid >> 5;
  const int  lane = tid & 31;
  const int  m    = lane & 15;
  const bool hi   = (lane & 16) != 0;
  const int  lrow = w * 16 + m;                   // 0..63 (padded neighbor idx)
  const int  le   = lrow < LNEI ? lrow : LNEI - 1;

  // ---- stage x = concat(e_uv, e_r) rows into LDS
  {
    const int*   hist  = hi ? history_r : history_uv;
    const float* table = hi ? r2e : v2e;
    int idx = hist[b * LNEI + le];
    const float* src = table + (size_t)idx * DD;
    float*       dst = xs + lrow * XS + (hi ? 64 : 0);
#if defined(HAVE_ASYNC_LDS)
    // CDNA5 async HBM->LDS: no VGPR round-trip, tracked on ASYNCcnt
#pragma unroll
    for (int i = 0; i < 16; ++i)
      __builtin_amdgcn_global_load_async_to_lds_b128(
          (v4i_t*)(src + i * 4), (v4i_t*)(dst + i * 4), 0, 0);
#else
    const float4* s4 = (const float4*)src;
    float4*       d4 = (float4*)dst;
#pragma unroll
    for (int i = 0; i < 16; ++i) d4[i] = s4[i];
#endif
  }
  if (tid < DD) uv_sh[tid] = u2e[(size_t)nodes[b] * DD + tid];
  wait_async0();
  __syncthreads();

  float*       xrow    = xs + lrow * XS;   // this lane's A-row base
  float*       wtile   = xs + (w * 16) * XS;
  float*       otile   = oh + (w * 16) * OS;
  const float* my_orow = oh + lrow * OS;

  // layer1: h1 = relu(x @ w_r1^T + b1)    xs[:,0:128] -> xs[:,0:64]
  gemm16x64<128, WT>(xrow, xrow + 64, w1w, w1b, wtile, XS, 0, lane);
  // layer2: o  = relu(h1 @ w_r2^T + b2)   xs[:,0:64]  -> o_hist rows
  gemm16x64<64, WT>(xrow, xrow, w2w, w2b, otile, OS, 0, lane);
  // att1: a1 = relu(concat(o, uv) @ att1^T + b)  -> xs[:,0:64]
  gemm16x64<128, WT>(my_orow, uv_sh, a1w, a1b, wtile, XS, 0, lane);
  // att2: a2 = relu(a1 @ att2^T + b)             -> xs[:,64:128]
  gemm16x64<64, WT>(xrow, xrow, a2w, a2b, wtile, XS, 64, lane);

  // att3: logits[l] = a2[l] . a3w + a3b  (pair of lanes splits the 64-dot)
  {
    const float* a2row = xrow + 64;
    const int kb = hi ? 32 : 0;
    float part = 0.f;
#pragma unroll
    for (int k = 0; k < 32; ++k) part += a2row[kb + k] * a3w[kb + k];
    part += __shfl_xor(part, 16, 32);
    if (lane < 16) lgt[lrow] = (lrow < LNEI) ? (part + a3b[0]) : -INFINITY;
  }
  __syncthreads();

  // softmax over neighbors + weighted pooling; one thread per output channel
  if (tid < DD) {
    float mx = -INFINITY;
#pragma unroll 1
    for (int l = 0; l < LNEI; ++l) mx = fmaxf(mx, lgt[l]);
    float s = 0.f, acc = 0.f;
#pragma unroll 1
    for (int l = 0; l < LNEI; ++l) {
      float e = __expf(lgt[l] - mx);
      s   += e;
      acc += e * oh[l * OS + tid];
    }
    out[(size_t)b * DD + tid] = acc / s;
  }
}

extern "C" void kernel_launch(void* const* d_in, const int* in_sizes, int n_in,
                              void* d_out, int out_size, void* d_ws, size_t ws_size,
                              hipStream_t stream) {
  const int*   nodes      = (const int*)d_in[0];
  const int*   history_uv = (const int*)d_in[1];
  const int*   history_r  = (const int*)d_in[2];
  const float* v2e        = (const float*)d_in[3];
  const float* u2e        = (const float*)d_in[4];
  const float* r2e        = (const float*)d_in[5];
  const float* w1w        = (const float*)d_in[6];
  const float* w1b        = (const float*)d_in[7];
  const float* w2w        = (const float*)d_in[8];
  const float* w2b        = (const float*)d_in[9];
  const float* a1w        = (const float*)d_in[10];
  const float* a1b        = (const float*)d_in[11];
  const float* a2w        = (const float*)d_in[12];
  const float* a2b        = (const float*)d_in[13];
  const float* a3w        = (const float*)d_in[14];
  const float* a3b        = (const float*)d_in[15];

  const int B = in_sizes[0];   // 8192

  if (ws_size >= (size_t)WH_TOTAL * 2) {
    _Float16* wh = (_Float16*)d_ws;
    cvt_weights_kernel<<<dim3(WH_TOTAL / 256), dim3(256), 0, stream>>>(
        w1w, w2w, a1w, a2w, wh);
    uv_agg_kernel<_Float16><<<dim3(B), dim3(128), 0, stream>>>(
        nodes, history_uv, history_r, v2e, u2e, r2e,
        wh, w1b, wh + 8192, w2b, wh + 12288, a1b, wh + 20480, a2b,
        a3w, a3b, (float*)d_out);
  } else {
    uv_agg_kernel<float><<<dim3(B), dim3(128), 0, stream>>>(
        nodes, history_uv, history_r, v2e, u2e, r2e,
        w1w, w1b, w2w, w2b, a1w, a1b, a2w, a2b,
        a3w, a3b, (float*)d_out);
  }
}
